// COM_HGNN_K4_55035710931169
// MI455X (gfx1250) — matrix operand order repaired and checked by reading, not compiled, not run on hardware
//
#include <hip/hip_runtime.h>

// ---------------------------------------------------------------------------
// CDNA5 / gfx1250 hetero-GNN forward. All dense transforms run through
// v_wmma_f32_16x16x32_bf16 (fp32 accumulate). Scatter-add uses native
// global_atomic_add_f32. Features stay L2-resident (80MB << 192MB L2).
// ---------------------------------------------------------------------------

typedef __attribute__((ext_vector_type(16))) __bf16 v16bf;
typedef __attribute__((ext_vector_type(8)))  float  v8f;

#define TM 128
#define TN 128
#define TK 32
#define PADB 8                      // row stride 40 bf16 = 80B (16B aligned)
#define LROW (TK + PADB)

// ---------------------------------------------------------------------------
// Generic WMMA GEMM:  D = epi( A[M,K] @ W[N,K]^T  (+bias) )
// flags bit0: D += acc (read-modify-write D)   flags bit1: ReLU
// Workgroup = 256 threads = 8 wave32; block tile 128x128; wave tile 32x64.
// Requires K % 4 == 0 (true for 1800/900/512/300/128) and 16B-aligned A/W.
// ---------------------------------------------------------------------------
__global__ __launch_bounds__(256) void gemm_wmma_bf16(
    const float* __restrict__ A, const float* __restrict__ W,
    const float* __restrict__ bias, float* __restrict__ D,
    int M, int N, int K, int flags)
{
  __shared__ alignas(16) __bf16 sA[TM][LROW];   // [row][k]
  __shared__ alignas(16) __bf16 sB[TN][LROW];   // [n][k]  (B = W^T: W rows)

  const int tid  = threadIdx.x;
  const int lane = tid & 31;
  const int wave = tid >> 5;
  const int wm   = wave >> 1;     // 0..3  -> 32-row slice
  const int wn   = wave & 1;      // 0..1  -> 64-col slice
  const int m0   = blockIdx.x * TM;
  const int n0   = blockIdx.y * TN;
  const int row  = lane & 15;     // also C/D column index
  const int h    = lane >> 4;     // lane half

  v8f acc[2][4] = {};

  for (int k0 = 0; k0 < K; k0 += TK) {
    // prefetch next A K-chunk (global_prefetch_b8)
    if (k0 + TK < K) {
      int pr = m0 + (tid >> 1);
      if (pr < M) __builtin_prefetch(&A[(size_t)pr * K + k0 + TK], 0, 1);
    }
    // --- stage A tile (128x32): float4 loads, branch-free predication ---
#pragma unroll
    for (int t = 0; t < (TM * TK) / (256 * 4); ++t) {
      const int i  = (t * 256 + tid) * 4;
      const int r  = i >> 5, k = i & 31;
      const int gr = m0 + r, gk = k0 + k;
      const bool ok = (gr < M) && (gk + 3 < K);
      const float4* p =
          (const float4*)(A + (size_t)(ok ? gr : 0) * K + (ok ? gk : 0));
      float4 v = *p;
      if (!ok) v = make_float4(0.f, 0.f, 0.f, 0.f);
      union { __bf16 b[4]; uint2 u; } pk;
      pk.b[0] = (__bf16)v.x; pk.b[1] = (__bf16)v.y;
      pk.b[2] = (__bf16)v.z; pk.b[3] = (__bf16)v.w;
      *(uint2*)&sA[r][k] = pk.u;                    // ds_store_b64
    }
    // --- stage B tile (W rows, 128x32) ---
#pragma unroll
    for (int t = 0; t < (TN * TK) / (256 * 4); ++t) {
      const int i  = (t * 256 + tid) * 4;
      const int n  = i >> 5, k = i & 31;
      const int gn = n0 + n, gk = k0 + k;
      const bool ok = (gn < N) && (gk + 3 < K);
      const float4* p =
          (const float4*)(W + (size_t)(ok ? gn : 0) * K + (ok ? gk : 0));
      float4 v = *p;
      if (!ok) v = make_float4(0.f, 0.f, 0.f, 0.f);
      union { __bf16 b[4]; uint2 u; } pk;
      pk.b[0] = (__bf16)v.x; pk.b[1] = (__bf16)v.y;
      pk.b[2] = (__bf16)v.z; pk.b[3] = (__bf16)v.w;
      *(uint2*)&sB[n][k] = pk.u;
    }
    __syncthreads();

    // --- fragments in contiguous k order (k = 16h + s), same for A and B: two ds_load_b128 each ---
    union { v16bf v; uint4 q[2]; } af[2], bfm[4];
#pragma unroll
    for (int i = 0; i < 2; i++) {
      const uint4* ap = (const uint4*)&sA[wm * 32 + i * 16 + row][0];
      af[i].q[0] = ap[h * 2];      // K = 16h + 0..7
      af[i].q[1] = ap[h * 2 + 1];  // K = 16h + 8..15
    }
#pragma unroll
    for (int j = 0; j < 4; j++) {
      const uint4* bp = (const uint4*)&sB[wn * 64 + j * 16 + row][0];
      bfm[j].q[0] = bp[h * 2];     // K = 16h + 0..7
      bfm[j].q[1] = bp[h * 2 + 1]; // K = 16h + 8..15
    }

    // --- 8 WMMAs per K-chunk ---
#pragma unroll
    for (int i = 0; i < 2; i++)
#pragma unroll
      for (int j = 0; j < 4; j++)
        acc[i][j] = __builtin_amdgcn_wmma_f32_16x16x32_bf16(
            false, af[i].v, false, bfm[j].v, (short)0, acc[i][j], false, false);
    __syncthreads();
  }

  // --- epilogue: C/D layout lane = N col, VGPR r -> M = 8*h + r ---
  const bool full = (m0 + TM <= M) && (n0 + TN <= N);
  float bj[4];
#pragma unroll
  for (int j = 0; j < 4; j++) {
    const int gn = n0 + wn * 64 + j * 16 + row;
    bj[j] = (bias && gn < N) ? bias[gn] : 0.f;
  }
  if (full) {
#pragma unroll
    for (int i = 0; i < 2; i++) {
#pragma unroll
      for (int j = 0; j < 4; j++) {
        const int gn = n0 + wn * 64 + j * 16 + row;
#pragma unroll
        for (int r = 0; r < 8; r++) {
          const int gr = m0 + wm * 32 + i * 16 + h * 8 + r;
          float v = acc[i][j][r] + bj[j];
          if (flags & 2) v = fmaxf(v, 0.f);
          const size_t idx = (size_t)gr * N + gn;
          if (flags & 1) v += D[idx];
          D[idx] = v;
        }
      }
    }
  } else {
#pragma unroll
    for (int i = 0; i < 2; i++) {
#pragma unroll
      for (int j = 0; j < 4; j++) {
        const int gn = n0 + wn * 64 + j * 16 + row;
#pragma unroll
        for (int r = 0; r < 8; r++) {
          const int gr = m0 + wm * 32 + i * 16 + h * 8 + r;
          if (gr < M && gn < N) {
            float v = acc[i][j][r] + bj[j];
            if (flags & 2) v = fmaxf(v, 0.f);
            const size_t idx = (size_t)gr * N + gn;
            if (flags & 1) v += D[idx];
            D[idx] = v;
          }
        }
      }
    }
  }
}

// ---------------------------------------------------------------------------
// One wave per edge; each lane moves float4 and issues 4 fp32 global atomics.
// ei layout: [2,E] row-major -> ei[0..E) = src, ei[E..2E) = dst.
// ---------------------------------------------------------------------------
__global__ __launch_bounds__(256) void scatter_add_k(
    const float* __restrict__ xsrc, const int* __restrict__ ei, int E,
    float* __restrict__ agg, float* __restrict__ deg)
{
  const int gw   = (int)((blockIdx.x * 256u + threadIdx.x) >> 5);
  const int lane = threadIdx.x & 31;
  if (gw >= E) return;
  const int src = ei[gw];
  const int dst = ei[E + gw];
  const float4 v = *(const float4*)(xsrc + (size_t)src * 128 + lane * 4);
  float* d = agg + (size_t)dst * 128 + lane * 4;
  unsafeAtomicAdd(d + 0, v.x);
  unsafeAtomicAdd(d + 1, v.y);
  unsafeAtomicAdd(d + 2, v.z);
  unsafeAtomicAdd(d + 3, v.w);
  if (deg && lane == 0) unsafeAtomicAdd(deg + dst, 1.0f);
}

__global__ __launch_bounds__(256) void zero_f32_k(float4* __restrict__ p, size_t n4) {
  size_t i = (size_t)blockIdx.x * 256 + threadIdx.x;
  const size_t stride = (size_t)gridDim.x * 256;
  const float4 z = make_float4(0.f, 0.f, 0.f, 0.f);
  for (; i < n4; i += stride) p[i] = z;
}

__global__ __launch_bounds__(256) void mean_div_k(float4* __restrict__ agg,
                                                  const float* __restrict__ deg,
                                                  size_t n4) {
  size_t i = (size_t)blockIdx.x * 256 + threadIdx.x;
  if (i >= n4) return;
  const float s = 1.0f / fmaxf(deg[i >> 5], 1.0f);  // 32 float4 per 128-col row
  float4 v = agg[i];
  v.x *= s; v.y *= s; v.z *= s; v.w *= s;
  agg[i] = v;
}

__global__ __launch_bounds__(256) void add_relu_k(float4* __restrict__ x,
                                                  const float4* __restrict__ nw,
                                                  size_t n4) {
  size_t i = (size_t)blockIdx.x * 256 + threadIdx.x;
  if (i >= n4) return;
  const float4 a = nw[i];
  float4 b = x[i];
  b.x += fmaxf(a.x, 0.f); b.y += fmaxf(a.y, 0.f);
  b.z += fmaxf(a.z, 0.f); b.w += fmaxf(a.w, 0.f);
  x[i] = b;
}

__global__ __launch_bounds__(256) void apply_coeff_k(float* __restrict__ out,
                                                     const float* __restrict__ lin,
                                                     const float* __restrict__ ang,
                                                     int total) {
  int i = blockIdx.x * 256 + threadIdx.x;
  if (i >= total) return;
  const int c = i % 24, g = c / 6, j = c % 6;
  const float k = (j < 3) ? lin[g * 3 + j] : ang[g * 3 + (j - 3)];
  out[i] *= k;
}

// ---------------------------------------------------------------------------
extern "C" void kernel_launch(void* const* d_in, const int* in_sizes, int n_in,
                              void* d_out, int out_size, void* d_ws, size_t ws_size,
                              hipStream_t stream)
{
  const float* x_base_in  = (const float*)d_in[0];
  const float* x_joint_in = (const float*)d_in[1];
  const float* x_foot_in  = (const float*)d_in[2];
  const float* encWb = (const float*)d_in[3];
  const float* encbb = (const float*)d_in[4];
  const float* encWj = (const float*)d_in[5];
  const float* encbj = (const float*)d_in[6];
  const float* encWf = (const float*)d_in[7];
  const float* encbf = (const float*)d_in[8];
  const float* Wrel  = (const float*)d_in[9];   // [L,8,128,128]
  const float* Wroot = (const float*)d_in[10];  // [L,8,128,128]
  const float* brel  = (const float*)d_in[11];  // [L,8,128]
  const float* btW1  = (const float*)d_in[12];
  const float* btb1  = (const float*)d_in[13];
  const float* btW2  = (const float*)d_in[14];
  const float* btb2  = (const float*)d_in[15];
  const float* decW  = (const float*)d_in[16];  // [24, 512]
  const float* decb  = (const float*)d_in[17];
  const float* clin  = (const float*)d_in[18];
  const float* cang  = (const float*)d_in[19];

  const int NB = in_sizes[0] / 1800;
  const int NJ = in_sizes[1] / 300;
  const int NF = in_sizes[2] / 900;
  const int H = 128, Lnum = 4, NET = 8;

  // ---- workspace carve-up (fp32) ----
  float* ws = (float*)d_ws;
  size_t o = 0;
  float* xb  = ws + o; o += (size_t)NB * H;
  float* xj  = ws + o; o += (size_t)NJ * H;
  float* xf  = ws + o; o += (size_t)NF * H;
  float* nb  = ws + o; o += (size_t)NB * H;   // nb,nj,nf contiguous -> one zero
  float* nj  = ws + o; o += (size_t)NJ * H;
  float* nf  = ws + o; o += (size_t)NF * H;
  size_t maxN = (size_t)(NB > NJ ? NB : NJ); if ((size_t)NF > maxN) maxN = NF;
  float* agg = ws + o; o += maxN * H;         // also reused as bt scratch
  float* deg = ws + o; o += maxN;
  (void)ws_size;

  auto gemm = [&](const float* A, const float* Wt, const float* b, float* D,
                  int M, int N, int K, int flags) {
    dim3 g((unsigned)((M + TM - 1) / TM), (unsigned)((N + TN - 1) / TN));
    gemm_wmma_bf16<<<g, dim3(256), 0, stream>>>(A, Wt, b, D, M, N, K, flags);
  };
  auto zero = [&](float* p, size_t n) {      // n must be a multiple of 4
    size_t n4 = n / 4;
    unsigned b = (unsigned)((n4 + 255) / 256);
    if (b > 4096u) b = 4096u;
    if (b == 0u) b = 1u;
    zero_f32_k<<<b, 256, 0, stream>>>((float4*)p, n4);
  };

  // ---- encoder: x = relu(X @ W^T + b) ----
  gemm(x_base_in,  encWb, encbb, xb, NB, H, 1800, /*relu*/2);
  gemm(x_joint_in, encWj, encbj, xj, NJ, H, 300,  2);
  gemm(x_foot_in,  encWf, encbf, xf, NF, H, 900,  2);

  // ---- edge-type tables ----
  const int  st[8]   = {0, 0, 0, 0, 1, 1, 1, 2};  // src node type
  const int  dt[8]   = {0, 0, 0, 1, 0, 1, 2, 1};  // dst node type
  const bool mn[8]   = {true, true, false, false, false, false, false, false};
  float* xarr[3] = {xb, xj, xf};
  float* narr[3] = {nb, nj, nf};
  const int Narr[3] = {NB, NJ, NF};

  for (int l = 0; l < Lnum; ++l) {
    zero(nb, (size_t)(NB + NJ + NF) * H);  // contiguous new buffers
    for (int t = 0; t < NET; ++t) {
      const int* ei = (const int*)d_in[20 + t];
      const int  E  = in_sizes[20 + t] / 2;
      const int  Nd = Narr[dt[t]];
      zero(agg, (size_t)Nd * H);
      if (mn[t]) zero(deg, (size_t)Nd);
      {
        unsigned blks = (unsigned)((E + 7) / 8);  // 8 waves (edges) per block
        scatter_add_k<<<blks, 256, 0, stream>>>(xarr[st[t]], ei, E, agg,
                                                mn[t] ? deg : nullptr);
      }
      if (mn[t]) {
        size_t n4 = (size_t)Nd * H / 4;
        mean_div_k<<<(unsigned)((n4 + 255) / 256), 256, 0, stream>>>(
            (float4*)agg, deg, n4);
      }
      const size_t woff = ((size_t)l * NET + t) * H * H;
      const size_t boff = ((size_t)l * NET + t) * H;
      gemm(agg,          Wrel  + woff, brel + boff, narr[dt[t]], Nd, H, H, /*add*/1);
      gemm(xarr[dt[t]],  Wroot + woff, nullptr,     narr[dt[t]], Nd, H, H, 1);
    }
    // base transform: x_base = relu(new_b @ W1^T + b1) @ W2^T + b2 + x_base
    gemm(nb,  btW1, btb1, agg, NB, H, H, /*relu*/2);     // agg reused as scratch
    gemm(agg, btW2, btb2, xb,  NB, H, H, /*add*/1);
    // joint/foot residual: x = relu(new) + x
    {
      size_t n4 = (size_t)NJ * H / 4;
      add_relu_k<<<(unsigned)((n4 + 255) / 256), 256, 0, stream>>>(
          (float4*)xj, (const float4*)nj, n4);
      n4 = (size_t)NF * H / 4;
      add_relu_k<<<(unsigned)((n4 + 255) / 256), 256, 0, stream>>>(
          (float4*)xf, (const float4*)nf, n4);
    }
  }

  // ---- decoder: out[bs,24] = x_base.view(bs,512) @ dec_W^T + dec_b ----
  const int bs = NB / 4;
  gemm(xb, decW, decb, (float*)d_out, bs, 24, 4 * H, 0);
  {
    const int total = bs * 24;
    apply_coeff_k<<<(unsigned)((total + 255) / 256), 256, 0, stream>>>(
        (float*)d_out, clin, cang, total);
  }
}
